// MinkowskiInverseSparseBottleneckV2_70282844832266
// MI455X (gfx1250) — matrix operand order repaired and checked by reading, not compiled, not run on hardware
//
#include <hip/hip_runtime.h>

typedef __bf16 bf16;
typedef __attribute__((ext_vector_type(16))) __bf16 v16bf;
typedef __attribute__((ext_vector_type(8)))  float  v8f;

#define N_PTS  200000
#define CIN    256
#define CMID   64
#define KOFF   9
#define NTILES (N_PTS / 16)   // 12500, exact

__device__ __forceinline__ float bnrelu(float v, float s, float t) {
    float r = fmaf(v, s, t);
    return r > 0.f ? r : 0.f;
}

// Fill 8 A-fragment slots [slot0, slot0+8) from 8 consecutive channels starting
// at channel c of row `src`, applying per-channel scale/shift + ReLU, times mask m.
// Matches CDNA5 16-bit A layout: slots are consecutive K within a run.
__device__ __forceinline__ void pack8(v16bf& a, int slot0, const float* src,
                                      const float* sS, const float* sT,
                                      int c, float m) {
    const float4 p0 = *(const float4*)(src + c);
    const float4 p1 = *(const float4*)(src + c + 4);
    const float4 s0 = *(const float4*)(sS + c);
    const float4 s1 = *(const float4*)(sS + c + 4);
    const float4 t0 = *(const float4*)(sT + c);
    const float4 t1 = *(const float4*)(sT + c + 4);
    a[slot0 + 0] = (bf16)(bnrelu(p0.x, s0.x, t0.x) * m);
    a[slot0 + 1] = (bf16)(bnrelu(p0.y, s0.y, t0.y) * m);
    a[slot0 + 2] = (bf16)(bnrelu(p0.z, s0.z, t0.z) * m);
    a[slot0 + 3] = (bf16)(bnrelu(p0.w, s0.w, t0.w) * m);
    a[slot0 + 4] = (bf16)(bnrelu(p1.x, s1.x, t1.x) * m);
    a[slot0 + 5] = (bf16)(bnrelu(p1.y, s1.y, t1.y) * m);
    a[slot0 + 6] = (bf16)(bnrelu(p1.z, s1.z, t1.z) * m);
    a[slot0 + 7] = (bf16)(bnrelu(p1.w, s1.w, t1.w) * m);
}

// ---------------- small utility kernels ----------------

__global__ void k_init(float* stats) {
    int i = threadIdx.x;
    if (i < 768) stats[i] = 0.f;  // sum0,sq0 (256 each) + sum1,sq1,sum2,sq2 (64 each)
}

// Transposed bf16 weights: W1T[n*256+k]=W1[k*64+n]; W2T[kk][d*64+c]=W2[kk][c*64+d];
// W3T[n*64+k]=W3[k*256+n]
__global__ void k_wprep(const float* __restrict__ W1, const float* __restrict__ W2,
                        const float* __restrict__ W3,
                        bf16* __restrict__ W1T, bf16* __restrict__ W2T,
                        bf16* __restrict__ W3T) {
    int i = blockIdx.x * 256 + threadIdx.x;
    if (i < CMID * CIN) {                 // 16384
        int n = i >> 8, k = i & 255;
        W1T[i] = (bf16)W1[k * CMID + n];
    }
    if (i < KOFF * CMID * CMID) {         // 36864
        int kk = i >> 12, r = i & 4095, d = r >> 6, c = r & 63;
        W2T[i] = (bf16)W2[kk * 4096 + c * CMID + d];
    }
    if (i < CIN * CMID) {                 // 16384
        int n = i >> 6, k = i & 63;
        W3T[i] = (bf16)W3[k * CIN + n];
    }
}

// per-channel sum / sumsq over [n, C] (C = 64 or 256, power of 2)
__global__ __launch_bounds__(256) void k_stats(const float* __restrict__ in, int n,
                                               int C, float* sum, float* sq) {
    __shared__ float ls[256], lq[256];
    int tid = threadIdx.x;
    int c = tid & (C - 1);
    int g = tid / C;
    int ng = 256 / C;
    float s = 0.f, q = 0.f;
    for (int row = blockIdx.x * ng + g; row < n; row += gridDim.x * ng) {
        float v = in[(size_t)row * C + c];
        s += v; q += v * v;
    }
    ls[tid] = s; lq[tid] = q;
    __syncthreads();
    if (tid < C) {
        for (int gg = 1; gg < ng; ++gg) { s += ls[tid + gg * C]; q += lq[tid + gg * C]; }
        atomicAdd(&sum[c], s);
        atomicAdd(&sq[c], q);
    }
}

__global__ void k_finalize(const float* sum, const float* sq, const float* g,
                           const float* be, int n, int C, float* scale, float* shift) {
    int c = threadIdx.x;
    if (c >= C) return;
    float inv_n = 1.f / (float)n;
    float mean = sum[c] * inv_n;
    float var  = sq[c] * inv_n - mean * mean;   // biased var, matches jnp.var
    float sc = g[c] * rsqrtf(var + 1e-5f);
    scale[c] = sc;
    shift[c] = be[c] - mean * sc;
}

// ---------------- GEMM1: h1 = relu(bn0(x)) @ W1 + b1  [N,64] ----------------

__global__ __launch_bounds__(256) void k_gemm1(const float* __restrict__ x,
                                               const float* __restrict__ b1,
                                               const float* __restrict__ scale0,
                                               const float* __restrict__ shift0,
                                               const bf16* __restrict__ W1T,
                                               float* __restrict__ h1) {
    __shared__ bf16 sW[CMID * CIN];          // 32 KB
    __shared__ float sS[CIN], sT[CIN];
    int tid = threadIdx.x;
    for (int i = tid; i < (CMID * CIN) / 8; i += 256)
        ((float4*)sW)[i] = ((const float4*)W1T)[i];
    if (tid < CIN) { sS[tid] = scale0[tid]; sT[tid] = shift0[tid]; }
    __syncthreads();

    int w = tid >> 5;
    int tile = blockIdx.x * 8 + w;
    if (tile >= NTILES) return;
    int lane = tid & 31, lo = lane & 15, hi = lane >> 4;
    int row0 = tile * 16;
    const float* xr = x + (size_t)(row0 + lo) * CIN;

    v8f acc[4] = {};
    #pragma unroll
    for (int kb = 0; kb < CIN; kb += 32) {
        __builtin_prefetch(xr + kb + 32, 0, 1);
        v16bf a;
        pack8(a, 0, xr, sS, sT, kb + hi * 16, 1.f);
        pack8(a, 8, xr, sS, sT, kb + hi * 16 + 8, 1.f);
        #pragma unroll
        for (int ct = 0; ct < 4; ++ct) {
            v16bf b = *(const v16bf*)(&sW[(ct * 16 + lo) * CIN + kb + hi * 16]);
            acc[ct] = __builtin_amdgcn_wmma_f32_16x16x32_bf16(
                false, a, false, b, (short)0, acc[ct], false, false);
        }
    }
    #pragma unroll
    for (int ct = 0; ct < 4; ++ct) {
        int col = ct * 16 + lo;
        float bias = b1[col];
        #pragma unroll
        for (int r = 0; r < 8; ++r)
            h1[(size_t)(row0 + r + hi * 8) * CMID + col] = acc[ct][r] + bias;
    }
}

// ------- sparse conv: h2[n] = sum_k gather(relu(bn1(h1)), idx[k]) @ W2[k] + b2 -------

__global__ __launch_bounds__(256) void k_conv(const float* __restrict__ h1,
                                              const int* __restrict__ idx,
                                              const float* __restrict__ b2,
                                              const float* __restrict__ scale1,
                                              const float* __restrict__ shift1,
                                              const bf16* __restrict__ W2T,
                                              float* __restrict__ h2) {
    __shared__ bf16 sW[CMID * CMID];         // 8 KB, one k-offset at a time
    __shared__ float sS[CMID], sT[CMID];
    int tid = threadIdx.x;
    if (tid < CMID) { sS[tid] = scale1[tid]; sT[tid] = shift1[tid]; }
    int w = tid >> 5;
    int tile = blockIdx.x * 8 + w;
    bool active = tile < NTILES;
    int lane = tid & 31, lo = lane & 15, hi = lane >> 4;
    int row0 = tile * 16;

    v8f acc[4] = {};
    for (int ko = 0; ko < KOFF; ++ko) {
        __syncthreads();
        for (int i = tid; i < (CMID * CMID) / 8; i += 256)
            ((float4*)sW)[i] = ((const float4*)(W2T + (size_t)ko * CMID * CMID))[i];
        __syncthreads();
        if (active) {
            if (ko + 1 < KOFF)
                __builtin_prefetch(idx + (size_t)(ko + 1) * N_PTS + row0 + lo, 0, 1);
            int ridx = idx[(size_t)ko * N_PTS + row0 + lo];
            bool valid = (unsigned)ridx < (unsigned)N_PTS;   // sentinel == N -> zero row
            float m = valid ? 1.f : 0.f;
            const float* hr = h1 + (size_t)(valid ? ridx : 0) * CMID;
            #pragma unroll
            for (int kb = 0; kb < CMID; kb += 32) {
                v16bf a;
                pack8(a, 0, hr, sS, sT, kb + hi * 16, m);
                pack8(a, 8, hr, sS, sT, kb + hi * 16 + 8, m);
                #pragma unroll
                for (int ct = 0; ct < 4; ++ct) {
                    v16bf b = *(const v16bf*)(&sW[(ct * 16 + lo) * CMID + kb + hi * 16]);
                    acc[ct] = __builtin_amdgcn_wmma_f32_16x16x32_bf16(
                        false, a, false, b, (short)0, acc[ct], false, false);
                }
            }
        }
    }
    if (active) {
        #pragma unroll
        for (int ct = 0; ct < 4; ++ct) {
            int col = ct * 16 + lo;
            float bias = b2[col];
            #pragma unroll
            for (int r = 0; r < 8; ++r)
                h2[(size_t)(row0 + r + hi * 8) * CMID + col] = acc[ct][r] + bias;
        }
    }
}

// ---- GEMM3 + residual: out = relu(bn2(h2)) @ W3 + b3 + relu(bn0(x))  [N,256] ----

__global__ __launch_bounds__(256) void k_gemm3(const float* __restrict__ h2,
                                               const float* __restrict__ x,
                                               const float* __restrict__ b3,
                                               const float* __restrict__ scale2,
                                               const float* __restrict__ shift2,
                                               const float* __restrict__ scale0,
                                               const float* __restrict__ shift0,
                                               const bf16* __restrict__ W3T,
                                               float* __restrict__ out) {
    __shared__ bf16 sW[CIN * CMID];          // 32 KB ; sW[n*64 + k]
    __shared__ float sS[CMID], sT[CMID], sS0[CIN], sT0[CIN];
    int tid = threadIdx.x;
    for (int i = tid; i < (CIN * CMID) / 8; i += 256)
        ((float4*)sW)[i] = ((const float4*)W3T)[i];
    if (tid < CMID) { sS[tid] = scale2[tid]; sT[tid] = shift2[tid]; }
    if (tid < CIN)  { sS0[tid] = scale0[tid]; sT0[tid] = shift0[tid]; }
    __syncthreads();

    int w = tid >> 5;
    int task = blockIdx.x * 8 + w;           // 4 column-groups of 64 per row tile
    if (task >= NTILES * 4) return;
    int tile = task >> 2, cg = task & 3;
    int lane = tid & 31, lo = lane & 15, hi = lane >> 4;
    int row0 = tile * 16;
    const float* hr = h2 + (size_t)(row0 + lo) * CMID;

    v8f acc[4] = {};
    #pragma unroll
    for (int kb = 0; kb < CMID; kb += 32) {
        v16bf a;
        pack8(a, 0, hr, sS, sT, kb + hi * 16, 1.f);
        pack8(a, 8, hr, sS, sT, kb + hi * 16 + 8, 1.f);
        #pragma unroll
        for (int ct = 0; ct < 4; ++ct) {
            int n = (cg * 4 + ct) * 16 + lo;
            v16bf b = *(const v16bf*)(&sW[n * CMID + kb + hi * 16]);
            acc[ct] = __builtin_amdgcn_wmma_f32_16x16x32_bf16(
                false, a, false, b, (short)0, acc[ct], false, false);
        }
    }
    #pragma unroll
    for (int ct = 0; ct < 4; ++ct) {
        int col = (cg * 4 + ct) * 16 + lo;
        float bias = b3[col];
        float sc = sS0[col], tc = sT0[col];
        #pragma unroll
        for (int r = 0; r < 8; ++r) {
            int row = row0 + r + hi * 8;
            float shortcut = bnrelu(x[(size_t)row * CIN + col], sc, tc);
            out[(size_t)row * CIN + col] = acc[ct][r] + bias + shortcut;
        }
    }
}

// ---------------- launch ----------------

extern "C" void kernel_launch(void* const* d_in, const int* in_sizes, int n_in,
                              void* d_out, int out_size, void* d_ws, size_t ws_size,
                              hipStream_t stream) {
    const float* x   = (const float*)d_in[0];
    const int*   idx = (const int*)d_in[1];
    const float* g0  = (const float*)d_in[2];
    const float* be0 = (const float*)d_in[3];
    const float* W1  = (const float*)d_in[4];
    const float* b1  = (const float*)d_in[5];
    const float* g1  = (const float*)d_in[6];
    const float* be1 = (const float*)d_in[7];
    const float* W2  = (const float*)d_in[8];
    const float* b2  = (const float*)d_in[9];
    const float* g2  = (const float*)d_in[10];
    const float* be2 = (const float*)d_in[11];
    const float* W3  = (const float*)d_in[12];
    const float* b3  = (const float*)d_in[13];
    float* out = (float*)d_out;

    char* ws = (char*)d_ws;
    bf16* W1T = (bf16*)(ws + 0);                 // 32768 B
    bf16* W2T = (bf16*)(ws + 32768);             // 73728 B
    bf16* W3T = (bf16*)(ws + 106496);            // 32768 B
    float* stats = (float*)(ws + 139264);        // 768 floats
    float* sum0 = stats;       float* sq0 = stats + 256;
    float* sum1 = stats + 512; float* sq1 = stats + 576;
    float* sum2 = stats + 640; float* sq2 = stats + 704;
    float* sc0 = (float*)(ws + 143360); float* sh0 = sc0 + 256;
    float* sc1 = sh0 + 256;             float* sh1 = sc1 + 64;
    float* sc2 = sh1 + 64;              float* sh2 = sc2 + 64;
    float* h1 = (float*)(ws + (1 << 20));
    float* h2 = (float*)(ws + (1 << 20) + (size_t)N_PTS * CMID * 4);

    k_init<<<1, 1024, 0, stream>>>(stats);
    k_wprep<<<144, 256, 0, stream>>>(W1, W2, W3, W1T, W2T, W3T);
    k_stats<<<512, 256, 0, stream>>>(x, N_PTS, CIN, sum0, sq0);
    k_finalize<<<1, 256, 0, stream>>>(sum0, sq0, g0, be0, N_PTS, CIN, sc0, sh0);
    k_gemm1<<<(NTILES + 7) / 8, 256, 0, stream>>>(x, b1, sc0, sh0, W1T, h1);
    k_stats<<<512, 256, 0, stream>>>(h1, N_PTS, CMID, sum1, sq1);
    k_finalize<<<1, 256, 0, stream>>>(sum1, sq1, g1, be1, N_PTS, CMID, sc1, sh1);
    k_conv<<<(NTILES + 7) / 8, 256, 0, stream>>>(h1, idx, b2, sc1, sh1, W2T, h2);
    k_stats<<<512, 256, 0, stream>>>(h2, N_PTS, CMID, sum2, sq2);
    k_finalize<<<1, 256, 0, stream>>>(sum2, sq2, g2, be2, N_PTS, CMID, sc2, sh2);
    k_gemm3<<<(NTILES * 4 + 7) / 8, 256, 0, stream>>>(h2, x, b3, sc2, sh2, sc0, sh0, W3T, out);
}
